// Model_Pre_40080634807207
// MI455X (gfx1250) — compile-verified, matrix operand order checked by reading, not run on hardware
//
#include <hip/hip_runtime.h>
#include <hip/hip_bf16.h>

typedef __attribute__((ext_vector_type(16))) _Float16 v16h;
typedef __attribute__((ext_vector_type(8)))  _Float16 v8h;
typedef __attribute__((ext_vector_type(8)))  float    v8f;

#define B_ 256
#define T_ 256
#define F_ 512
#define H_ 512

#define WMMA_F16(A, Bf, C) \
  __builtin_amdgcn_wmma_f32_16x16x32_f16(false, (A), false, (Bf), (short)0, (C), false, false)

// ---------------- fragment helpers (16x16x32 f16 WMMA) ----------------
// Row-major [rows][K] matrix; lane<16 reads K chunks {0..7,16..23}+k0,
// lane>=16 reads {8..15,24..31}+k0. p points at row*ld + k0 + koff.
__device__ __forceinline__ v16h frag_load(const _Float16* p) {
  v8h c0 = *(const v8h*)(p);
  v8h c1 = *(const v8h*)(p + 16);
  v16h a;
#pragma unroll
  for (int i = 0; i < 8; ++i) { a[i] = c0[i]; a[i + 8] = c1[i]; }
  return a;
}

__device__ __forceinline__ float sigmoidf_(float x) {
  return 1.0f / (1.0f + __expf(-x));
}

// ---------------- tiny prep kernels ----------------
__global__ void cvt_f16_k(const float* __restrict__ in, _Float16* __restrict__ out, int n) {
  int i = blockIdx.x * blockDim.x + threadIdx.x;
  if (i < n) out[i] = (_Float16)in[i];
}

// conv_w [o][f][d] (F,F,3) -> [d][o][f] f16
__global__ void conv_reorg_k(const float* __restrict__ cw, _Float16* __restrict__ out, int n) {
  int i = blockIdx.x * blockDim.x + threadIdx.x;
  if (i >= n) return;
  int d = i / (F_ * F_);
  int rem = i - d * (F_ * F_);
  int o = rem / F_, f = rem - o * F_;
  out[i] = (_Float16)cw[(size_t)o * F_ * 3 + f * 3 + d];
}

// x' padded: out[b, tt, f] for tt in [0,T+2); tt==0 / tt==T+1 are zero rows,
// else where(x==-1,0,x) in f16. Removes boundary predication from conv loads.
__global__ void xpad_k(const float* __restrict__ x, _Float16* __restrict__ out, int n) {
  for (int i = blockIdx.x * blockDim.x + threadIdx.x; i < n; i += gridDim.x * blockDim.x) {
    int f = i & (F_ - 1);
    int rem = i >> 9;                    // b*(T+2) + tt
    int tt = rem % (T_ + 2);
    int b  = rem / (T_ + 2);
    float v = 0.0f;
    if (tt >= 1 && tt <= T_) {
      v = x[((size_t)b * T_ + (tt - 1)) * F_ + f];
      v = (v == -1.0f) ? 0.0f : v;
    }
    out[i] = (_Float16)v;
  }
}

// time-since-last-observation scan over T per (b,f); coalesced in f
__global__ void time_last_k(const float* __restrict__ t, const float* __restrict__ mask,
                            _Float16* __restrict__ out) {
  int i = blockIdx.x * blockDim.x + threadIdx.x;   // over B*F
  if (i >= B_ * F_) return;
  int b = i / F_, f = i - b * F_;
  const float* tp = t + (size_t)b * T_ * F_ + f;
  const float* mp = mask + (size_t)b * T_ * F_ + f;
  _Float16*    op = out + (size_t)b * T_ * F_ + f;
  float prev = 0.0f;
  for (int tt = 0; tt < T_; ++tt) {
    float tm = tp[(size_t)tt * F_] * mp[(size_t)tt * F_];
    float tl = tm - prev;
    op[(size_t)tt * F_] = (_Float16)(tl > 0.0f ? tl : 0.0f);
    prev = prev > tm ? prev : tm;
  }
}

// ---------------- conv1d(k=3) as 3 accumulated WMMA GEMMs + relu ----------------
// out[b,t,o] = relu(bias[o] + sum_d sum_f xpad[b,t+d,f] * Wc[d][o][f])
// 48 segments, d-major (d = s>>4, k0 = (s&15)*32): shift-only address math.
// Ping-pong double-buffered (no register rotation).
__global__ __launch_bounds__(256) void conv_gemm_k(const _Float16* __restrict__ Xp,
                                                   const _Float16* __restrict__ Wc,
                                                   const float* __restrict__ bias,
                                                   _Float16* __restrict__ out) {
  int mt = blockIdx.x;                       // 4096 M-tiles of 16 rows
  int tid = threadIdx.x, lane = tid & 31, wv = tid >> 5;
  int r0 = lane & 15, koff = (lane >> 4) * 8, rb = (lane >> 4) * 8;
  int rowg = mt * 16 + r0;
  int b = rowg >> 8, t = rowg & 255;         // tiles never straddle batches (T=256)
  int nb = wv * 64;                          // wave covers 64 output cols

  const _Float16* xrow = Xp + ((size_t)(b * (T_ + 2) + t)) * F_ + koff;  // +d*F_ per d
  const _Float16* wrow = Wc + (size_t)(nb + r0) * F_ + koff;             // +d*F_*F_ +j*16*F_

  auto loadA = [&](int s) -> v16h {          // d = s>>4, k0 = (s&15)*32
    return frag_load(xrow + (s >> 4) * F_ + (s & 15) * 32);
  };
  auto loadB = [&](int s, int j) -> v16h {
    return frag_load(wrow + (size_t)(s >> 4) * (F_ * F_) + j * (16 * F_) + (s & 15) * 32);
  };

  v8f acc[4] = {};
  v16h a0 = loadA(0);
  v16h b00 = loadB(0, 0), b01 = loadB(0, 1), b02 = loadB(0, 2), b03 = loadB(0, 3);
  v16h a1, b10, b11, b12, b13;
#pragma unroll 1
  for (int s = 0; s < 46; s += 2) {
    a1  = loadA(s + 1);
    b10 = loadB(s + 1, 0); b11 = loadB(s + 1, 1); b12 = loadB(s + 1, 2); b13 = loadB(s + 1, 3);
    acc[0] = WMMA_F16(a0, b00, acc[0]);
    acc[1] = WMMA_F16(a0, b01, acc[1]);
    acc[2] = WMMA_F16(a0, b02, acc[2]);
    acc[3] = WMMA_F16(a0, b03, acc[3]);
    a0  = loadA(s + 2);
    b00 = loadB(s + 2, 0); b01 = loadB(s + 2, 1); b02 = loadB(s + 2, 2); b03 = loadB(s + 2, 3);
    acc[0] = WMMA_F16(a1, b10, acc[0]);
    acc[1] = WMMA_F16(a1, b11, acc[1]);
    acc[2] = WMMA_F16(a1, b12, acc[2]);
    acc[3] = WMMA_F16(a1, b13, acc[3]);
  }
  a1  = loadA(47);
  b10 = loadB(47, 0); b11 = loadB(47, 1); b12 = loadB(47, 2); b13 = loadB(47, 3);
  acc[0] = WMMA_F16(a0, b00, acc[0]);
  acc[1] = WMMA_F16(a0, b01, acc[1]);
  acc[2] = WMMA_F16(a0, b02, acc[2]);
  acc[3] = WMMA_F16(a0, b03, acc[3]);
  acc[0] = WMMA_F16(a1, b10, acc[0]);
  acc[1] = WMMA_F16(a1, b11, acc[1]);
  acc[2] = WMMA_F16(a1, b12, acc[2]);
  acc[3] = WMMA_F16(a1, b13, acc[3]);

#pragma unroll
  for (int j = 0; j < 4; ++j) {
    int col = nb + j * 16 + (lane & 15);
    float bv = bias[col];
#pragma unroll
    for (int r = 0; r < 8; ++r) {
      int row = mt * 16 + rb + r;
      float v = acc[j][r] + bv;
      out[(size_t)row * F_ + col] = (_Float16)(v > 0.0f ? v : 0.0f);
    }
  }
}

// ---------------- generic M x 512 x 512 GEMM: out = epi(A @ W^T + bias) ----------------
// mode 0: identity (xr/xz/xh).  mode 1: exp(-relu(.)) (gamma).  Ping-pong K-loop.
__global__ __launch_bounds__(256) void gemm512_k(const _Float16* __restrict__ A,
                                                 const _Float16* __restrict__ W,
                                                 const float* __restrict__ bias,
                                                 _Float16* __restrict__ out, int mode) {
  int mt = blockIdx.x;
  int tid = threadIdx.x, lane = tid & 31, wv = tid >> 5;
  int r0 = lane & 15, koff = (lane >> 4) * 8, rb = (lane >> 4) * 8;
  int nb = wv * 64;
  const _Float16* arow = A + (size_t)(mt * 16 + r0) * H_ + koff;
  const _Float16* wrow = W + (size_t)(nb + r0) * H_ + koff;   // +j*16*H_ immediates
  auto loadB = [&](int k0, int j) -> v16h {
    return frag_load(wrow + j * (16 * H_) + k0);
  };

  v8f acc[4] = {};
  v16h a0 = frag_load(arow);
  v16h b00 = loadB(0, 0), b01 = loadB(0, 1), b02 = loadB(0, 2), b03 = loadB(0, 3);
  v16h a1, b10, b11, b12, b13;
#pragma unroll 1
  for (int k0 = 0; k0 < H_ - 64; k0 += 64) {
    a1  = frag_load(arow + k0 + 32);
    b10 = loadB(k0 + 32, 0); b11 = loadB(k0 + 32, 1);
    b12 = loadB(k0 + 32, 2); b13 = loadB(k0 + 32, 3);
    acc[0] = WMMA_F16(a0, b00, acc[0]);
    acc[1] = WMMA_F16(a0, b01, acc[1]);
    acc[2] = WMMA_F16(a0, b02, acc[2]);
    acc[3] = WMMA_F16(a0, b03, acc[3]);
    a0  = frag_load(arow + k0 + 64);
    b00 = loadB(k0 + 64, 0); b01 = loadB(k0 + 64, 1);
    b02 = loadB(k0 + 64, 2); b03 = loadB(k0 + 64, 3);
    acc[0] = WMMA_F16(a1, b10, acc[0]);
    acc[1] = WMMA_F16(a1, b11, acc[1]);
    acc[2] = WMMA_F16(a1, b12, acc[2]);
    acc[3] = WMMA_F16(a1, b13, acc[3]);
  }
  a1  = frag_load(arow + H_ - 32);
  b10 = loadB(H_ - 32, 0); b11 = loadB(H_ - 32, 1);
  b12 = loadB(H_ - 32, 2); b13 = loadB(H_ - 32, 3);
  acc[0] = WMMA_F16(a0, b00, acc[0]);
  acc[1] = WMMA_F16(a0, b01, acc[1]);
  acc[2] = WMMA_F16(a0, b02, acc[2]);
  acc[3] = WMMA_F16(a0, b03, acc[3]);
  acc[0] = WMMA_F16(a1, b10, acc[0]);
  acc[1] = WMMA_F16(a1, b11, acc[1]);
  acc[2] = WMMA_F16(a1, b12, acc[2]);
  acc[3] = WMMA_F16(a1, b13, acc[3]);

#pragma unroll
  for (int j = 0; j < 4; ++j) {
    int col = nb + j * 16 + (lane & 15);
    float bv = bias[col];
#pragma unroll
    for (int r = 0; r < 8; ++r) {
      int row = mt * 16 + rb + r;
      float v = acc[j][r] + bv;
      if (mode == 1) v = __expf(-(v > 0.0f ? v : 0.0f));
      out[(size_t)row * H_ + col] = (_Float16)v;
    }
  }
}

// ---------------- GRU-D scan: 16 persistent WGs, each owns 16 batch rows ----------------
__global__ __launch_bounds__(256) void scan_k(const _Float16* __restrict__ gamma,
                                              const _Float16* __restrict__ xr,
                                              const _Float16* __restrict__ xz,
                                              const _Float16* __restrict__ xh,
                                              const _Float16* __restrict__ Whr,
                                              const _Float16* __restrict__ Whz,
                                              const _Float16* __restrict__ Whh,
                                              const int* __restrict__ slen,
                                              float* __restrict__ hlast) {
  __shared__ float    sh_h[16 * H_];     // 32 KB  hidden state (f32)
  __shared__ _Float16 sh_hd[16 * H_];    // 16 KB  gamma*h (f16 A-operand)
  __shared__ _Float16 sh_rhd[16 * H_];   // 16 KB  r*(gamma*h) (f16 A-operand)
  int tid = threadIdx.x, lane = tid & 31, wv = tid >> 5;
  int bb = blockIdx.x;                   // batch group, 16 rows
  int r0 = lane & 15, koff = (lane >> 4) * 8, rb = (lane >> 4) * 8;

  int myidx[8];                          // last-step index for my epilogue rows
#pragma unroll
  for (int r = 0; r < 8; ++r) myidx[r] = slen[bb * 16 + rb + r] - 1;

  for (int e = tid; e < 16 * H_; e += 256) sh_h[e] = 0.0f;
  __syncthreads();

#pragma unroll 1
  for (int t = 0; t < T_; ++t) {
    // hd = gamma_t * h
    for (int e = tid; e < 16 * H_; e += 256) {
      int row = e >> 9, col = e & (H_ - 1);
      float g = (float)gamma[((size_t)(bb * 16 + row) * T_ + t) * H_ + col];
      sh_hd[e] = (_Float16)(g * sh_h[e]);
    }
    __syncthreads();

    // phase A: r = sigmoid(xr + hd @ Whr^T); store r*hd (f16). Ping-pong K-loop.
#pragma unroll 1
    for (int i = 0; i < 4; ++i) {
      int nb = (wv * 4 + i) * 16;
      const _Float16* wrow = Whr + (size_t)(nb + r0) * H_ + koff;
      v8f acc = {};
      v16h a0 = frag_load(&sh_hd[r0 * H_ + koff]);
      v16h b0 = frag_load(wrow);
      v16h a1, b1;
#pragma unroll 1
      for (int k0 = 0; k0 < H_ - 64; k0 += 64) {
        a1 = frag_load(&sh_hd[r0 * H_ + k0 + 32 + koff]);
        b1 = frag_load(wrow + k0 + 32);
        acc = WMMA_F16(a0, b0, acc);
        a0 = frag_load(&sh_hd[r0 * H_ + k0 + 64 + koff]);
        b0 = frag_load(wrow + k0 + 64);
        acc = WMMA_F16(a1, b1, acc);
      }
      a1 = frag_load(&sh_hd[r0 * H_ + (H_ - 32) + koff]);
      b1 = frag_load(wrow + H_ - 32);
      acc = WMMA_F16(a0, b0, acc);
      acc = WMMA_F16(a1, b1, acc);

      int col = nb + (lane & 15);
#pragma unroll
      for (int r = 0; r < 8; ++r) {
        int row = rb + r;
        size_t gi = ((size_t)(bb * 16 + row) * T_ + t) * H_ + col;
        float rv = sigmoidf_(acc[r] + (float)xr[gi]);
        sh_rhd[row * H_ + col] = (_Float16)(rv * (float)sh_hd[row * H_ + col]);
      }
    }
    __syncthreads();

    // phase B: z = sigmoid(xz + hd@Whz^T), h~ = tanh(xh + (r*hd)@Whh^T). Ping-pong.
#pragma unroll 1
    for (int i = 0; i < 4; ++i) {
      int nb = (wv * 4 + i) * 16;
      const _Float16* wzrow = Whz + (size_t)(nb + r0) * H_ + koff;
      const _Float16* whrow = Whh + (size_t)(nb + r0) * H_ + koff;
      v8f accz = {}, acct = {};
      v16h ah0 = frag_load(&sh_hd[r0 * H_ + koff]);
      v16h ar0 = frag_load(&sh_rhd[r0 * H_ + koff]);
      v16h bz0 = frag_load(wzrow);
      v16h bh0 = frag_load(whrow);
      v16h ah1, ar1, bz1, bh1;
#pragma unroll 1
      for (int k0 = 0; k0 < H_ - 64; k0 += 64) {
        ah1 = frag_load(&sh_hd[r0 * H_ + k0 + 32 + koff]);
        ar1 = frag_load(&sh_rhd[r0 * H_ + k0 + 32 + koff]);
        bz1 = frag_load(wzrow + k0 + 32);
        bh1 = frag_load(whrow + k0 + 32);
        accz = WMMA_F16(ah0, bz0, accz);
        acct = WMMA_F16(ar0, bh0, acct);
        ah0 = frag_load(&sh_hd[r0 * H_ + k0 + 64 + koff]);
        ar0 = frag_load(&sh_rhd[r0 * H_ + k0 + 64 + koff]);
        bz0 = frag_load(wzrow + k0 + 64);
        bh0 = frag_load(whrow + k0 + 64);
        accz = WMMA_F16(ah1, bz1, accz);
        acct = WMMA_F16(ar1, bh1, acct);
      }
      ah1 = frag_load(&sh_hd[r0 * H_ + (H_ - 32) + koff]);
      ar1 = frag_load(&sh_rhd[r0 * H_ + (H_ - 32) + koff]);
      bz1 = frag_load(wzrow + H_ - 32);
      bh1 = frag_load(whrow + H_ - 32);
      accz = WMMA_F16(ah0, bz0, accz);
      acct = WMMA_F16(ar0, bh0, acct);
      accz = WMMA_F16(ah1, bz1, accz);
      acct = WMMA_F16(ar1, bh1, acct);

      int col = nb + (lane & 15);
#pragma unroll
      for (int r = 0; r < 8; ++r) {
        int row = rb + r;
        size_t gi = ((size_t)(bb * 16 + row) * T_ + t) * H_ + col;
        float zv = sigmoidf_(accz[r] + (float)xz[gi]);
        float ht = tanhf(acct[r] + (float)xh[gi]);
        float hd = (float)sh_hd[row * H_ + col];
        float hn = (1.0f - zv) * hd + zv * ht;
        sh_h[row * H_ + col] = hn;
        if (myidx[r] == t) hlast[(size_t)(bb * 16 + row) * H_ + col] = hn;
      }
    }
    __syncthreads();
  }
}

// ---------------- final: out[b,:] = (h_last + x_conv[b,idx]) @ W_pre^T + b_pre ----------------
__global__ __launch_bounds__(128) void final_k(const float* __restrict__ hlast,
                                               const _Float16* __restrict__ xconv,
                                               const int* __restrict__ slen,
                                               const float* __restrict__ Wpre,
                                               const float* __restrict__ bpre,
                                               float* __restrict__ out) {
  __shared__ float s0[128], s1[128];
  int b = blockIdx.x, tid = threadIdx.x;
  int idx = slen[b] - 1;
  const float*    hr = hlast + (size_t)b * H_;
  const _Float16* xc = xconv + ((size_t)b * T_ + idx) * F_;
  float a0 = 0.0f, a1 = 0.0f;
  for (int k = tid; k < H_; k += 128) {
    float v = hr[k] + (float)xc[k];
    a0 += v * Wpre[k];
    a1 += v * Wpre[H_ + k];
  }
  s0[tid] = a0; s1[tid] = a1;
  __syncthreads();
  for (int s = 64; s > 0; s >>= 1) {
    if (tid < s) { s0[tid] += s0[tid + s]; s1[tid] += s1[tid + s]; }
    __syncthreads();
  }
  if (tid == 0) {
    out[b * 2 + 0] = s0[0] + bpre[0];
    out[b * 2 + 1] = s1[0] + bpre[1];
  }
}

// ---------------- host launch ----------------
extern "C" void kernel_launch(void* const* d_in, const int* in_sizes, int n_in,
                              void* d_out, int out_size, void* d_ws, size_t ws_size,
                              hipStream_t stream) {
  const float* x      = (const float*)d_in[0];
  const float* mask   = (const float*)d_in[1];
  const float* tarr   = (const float*)d_in[2];
  const int*   slen   = (const int*)d_in[3];
  const float* conv_w = (const float*)d_in[4];
  const float* conv_b = (const float*)d_in[5];
  const float* W_dg   = (const float*)d_in[6];
  const float* b_dg   = (const float*)d_in[7];
  const float* W_xr   = (const float*)d_in[8];
  const float* b_xr   = (const float*)d_in[9];
  const float* W_hr   = (const float*)d_in[10];
  const float* W_xz   = (const float*)d_in[11];
  const float* b_xz   = (const float*)d_in[12];
  const float* W_hz   = (const float*)d_in[13];
  const float* W_xh   = (const float*)d_in[14];
  const float* b_xh   = (const float*)d_in[15];
  const float* W_hh   = (const float*)d_in[16];
  const float* W_pre  = (const float*)d_in[17];
  const float* b_pre  = (const float*)d_in[18];
  float* out = (float*)d_out;

  const size_t EBIG = (size_t)B_ * T_ * F_;        // 33,554,432
  const size_t EPAD = (size_t)B_ * (T_ + 2) * F_;  // 33,816,576
  char* p = (char*)d_ws;
  _Float16* xpad16   = (_Float16*)p; p += EPAD * 2;
  _Float16* xconv16  = (_Float16*)p; p += EBIG * 2;
  _Float16* tlast16  = (_Float16*)p; p += EBIG * 2;
  _Float16* gamma16  = (_Float16*)p; p += EBIG * 2;
  _Float16* xr16     = (_Float16*)p; p += EBIG * 2;
  _Float16* xz16     = (_Float16*)p; p += EBIG * 2;
  _Float16* xh16     = (_Float16*)p; p += EBIG * 2;
  _Float16* wc16     = (_Float16*)p; p += (size_t)3 * F_ * F_ * 2;
  _Float16* wdg16    = (_Float16*)p; p += (size_t)H_ * F_ * 2;
  _Float16* wxr16    = (_Float16*)p; p += (size_t)H_ * F_ * 2;
  _Float16* whr16    = (_Float16*)p; p += (size_t)H_ * H_ * 2;
  _Float16* wxz16    = (_Float16*)p; p += (size_t)H_ * F_ * 2;
  _Float16* whz16    = (_Float16*)p; p += (size_t)H_ * H_ * 2;
  _Float16* wxh16    = (_Float16*)p; p += (size_t)H_ * F_ * 2;
  _Float16* whh16    = (_Float16*)p; p += (size_t)H_ * H_ * 2;
  float*    hlast    = (float*)p;    p += (size_t)B_ * H_ * 4;

  const int WN = H_ * F_;                          // 262144
  // weight conversions
  cvt_f16_k<<<WN / 256, 256, 0, stream>>>(W_dg, wdg16, WN);
  cvt_f16_k<<<WN / 256, 256, 0, stream>>>(W_xr, wxr16, WN);
  cvt_f16_k<<<WN / 256, 256, 0, stream>>>(W_hr, whr16, WN);
  cvt_f16_k<<<WN / 256, 256, 0, stream>>>(W_xz, wxz16, WN);
  cvt_f16_k<<<WN / 256, 256, 0, stream>>>(W_hz, whz16, WN);
  cvt_f16_k<<<WN / 256, 256, 0, stream>>>(W_xh, wxh16, WN);
  cvt_f16_k<<<WN / 256, 256, 0, stream>>>(W_hh, whh16, WN);
  conv_reorg_k<<<(3 * F_ * F_) / 256, 256, 0, stream>>>(conv_w, wc16, 3 * F_ * F_);

  // padded x' and time_last
  xpad_k<<<4096, 256, 0, stream>>>(x, xpad16, (int)EPAD);
  time_last_k<<<(B_ * F_) / 256, 256, 0, stream>>>(tarr, mask, tlast16);

  // conv block (WMMA)
  conv_gemm_k<<<(B_ * T_) / 16, 256, 0, stream>>>(xpad16, wc16, conv_b, xconv16);

  // big GEMMs (WMMA): gamma + three input projections
  gemm512_k<<<(B_ * T_) / 16, 256, 0, stream>>>(tlast16, wdg16, b_dg, gamma16, 1);
  gemm512_k<<<(B_ * T_) / 16, 256, 0, stream>>>(xconv16, wxr16, b_xr, xr16, 0);
  gemm512_k<<<(B_ * T_) / 16, 256, 0, stream>>>(xconv16, wxz16, b_xz, xz16, 0);
  gemm512_k<<<(B_ * T_) / 16, 256, 0, stream>>>(xconv16, wxh16, b_xh, xh16, 0);

  // recurrent scan (WMMA, persistent per batch-group)
  scan_k<<<B_ / 16, 256, 0, stream>>>(gamma16, xr16, xz16, xh16,
                                      whr16, whz16, whh16, slen, hlast);

  // gather + residual + final projection
  final_k<<<B_, 128, 0, stream>>>(hlast, xconv16, slen, W_pre, b_pre, out);
}
